// singleSANNLayer_5712306504312
// MI455X (gfx1250) — matrix operand order repaired and checked by reading, not compiled, not run on hardware
//
#include <hip/hip_runtime.h>

// ---------------------------------------------------------------------------
// SANN hetero-graph attention layer for MI455X (gfx1250, wave32, WMMA).
// exp(a_src+a_dst-M)·mask factorizes; dst factor cancels in ret/dst_sum, so
// the per-head 'hnm,hne->hme' einsum collapses to one bf16 mask-GEMM per edge
// type shared across all 8 heads:
//     C(3072x272) = adjT(3072x3072, {0,1}) @ B(3072x[256 gfp | 8 g | pad])
// adj(0/1) is exact in bf16; accumulate f32 via v_wmma_f32_16x16x32_bf16.
// B is pre-swizzled into WMMA fragment order so the GEMM's B staging is an
// identity copy done with global_load_async_to_lds_b128 (ASYNCcnt pipeline).
// ---------------------------------------------------------------------------

typedef __attribute__((ext_vector_type(16))) __bf16 v16bf;
typedef __attribute__((ext_vector_type(8)))  float  v8f;

#define NN    3072     // nodes per set
#define DIN   256
#define NH    8        // heads
#define DH    32       // head dim
#define BC    272      // C columns: 256 gfp + 8 g + 8 pad
#define NKT   96       // K tiles (3072/32)
#define NCT   18       // B col-tiles of 16 (288 cols, zero past 264)
#define CTPG  6        // col-tiles per workgroup (3 groups)

union FragBF { v16bf v; uint4 q[2]; };

__device__ __forceinline__ unsigned short f32_to_bf16(float f) {
    unsigned int u = __float_as_uint(f);
    unsigned int r = u + 0x7FFFu + ((u >> 16) & 1u);   // round-to-nearest-even
    return (unsigned short)(r >> 16);
}

// ---------------------------------------------------------------------------
// Kernel 1: fp[h,n,e] = emb_f[n,:] @ W[f,t,h,:,e];  a_src[h,n] = fp . attn
// grid (2, 8, 192), block 256.
// ---------------------------------------------------------------------------
__global__ __launch_bounds__(256) void proj_kernel(
    const float* __restrict__ emb0, const float* __restrict__ emb1,
    const float* __restrict__ W, const float* __restrict__ attn,
    float* __restrict__ fp_out, float* __restrict__ a_out)
{
    __shared__ float Wl[DIN][DH];   // 32 KB
    __shared__ float El[16][DIN];   // 16 KB
    const int i  = blockIdx.x;              // edge type
    const int h  = blockIdx.y;
    const int n0 = blockIdx.z * 16;
    const int tid = threadIdx.x;

    const int f = i, t = 1 - i;
    const float* emb = (f == 0) ? emb0 : emb1;
    const float* Wg  = W + (size_t)((f*2 + t)*NH + h) * (DIN*DH);

    #pragma unroll
    for (int p = 0; p < 32; ++p) {
        int idx = tid + (p << 8);
        Wl[idx >> 5][idx & 31] = Wg[idx];
    }
    #pragma unroll
    for (int p = 0; p < 16; ++p) {
        int idx = tid + (p << 8);
        El[idx >> 8][idx & 255] = emb[(size_t)(n0 + (idx >> 8))*DIN + (idx & 255)];
    }
    __syncthreads();

    const int e = tid & 31;                 // lane within wave32
    const float av = attn[(h*2*DH + e)*2 + i];
    #pragma unroll
    for (int pass = 0; pass < 2; ++pass) {
        const int nn = (tid >> 5) + pass*8;
        float acc = 0.f;
        #pragma unroll 8
        for (int d = 0; d < DIN; ++d) acc += El[nn][d] * Wl[d][e];
        fp_out[((size_t)(i*NH + h)*NN + n0 + nn)*DH + e] = acc;
        float pa = acc * av;
        #pragma unroll
        for (int off = 16; off > 0; off >>= 1) pa += __shfl_xor(pa, off);
        if (e == 0) a_out[(size_t)(i*NH + h)*NN + n0 + nn] = pa;
    }
}

// ---------------------------------------------------------------------------
// Kernel 2: per-(edge-type, head) max of a_src -> c1.  grid 16, block 256.
// ---------------------------------------------------------------------------
__global__ __launch_bounds__(256) void max_kernel(
    const float* __restrict__ a_src, float* __restrict__ c1)
{
    __shared__ float red[256];
    const int b = blockIdx.x;
    const int tid = threadIdx.x;
    float m = -3.0e38f;
    for (int idx = tid; idx < NN; idx += 256)
        m = fmaxf(m, a_src[(size_t)b*NN + idx]);
    red[tid] = m;
    __syncthreads();
    for (int s = 128; s > 0; s >>= 1) {
        if (tid < s) red[tid] = fmaxf(red[tid], red[tid + s]);
        __syncthreads();
    }
    if (tid == 0) c1[b] = red[0];
}

// ---------------------------------------------------------------------------
// Kernel 3: build B in WMMA *fragment order*:
//   Bfrag[((i*NKT+kt)*NCT + ctf)*32 + lane][16 ushorts]
// element k (0..15) of a lane = (kk = ((lane&16)>>1)+(k&7)+((k&8)<<1), col = ctf*16 + (lane&15)).
// Each thread writes one lane's 32 bytes -> fully coalesced b128 stores.
// grid 432, block 256 (2*96*18*32 threads).
// ---------------------------------------------------------------------------
__global__ __launch_bounds__(256) void bmat_kernel(
    const float* __restrict__ fp, const float* __restrict__ a_src,
    const float* __restrict__ c1, unsigned short* __restrict__ Bfrag)
{
    const int g    = blockIdx.x*256 + threadIdx.x;
    const int lane = g & 31;
    const int ctf  = (g >> 5) % NCT;
    const int kt   = ((g >> 5) / NCT) % NKT;
    const int i    = g / (32*NCT*NKT);
    const int col  = ctf*16 + (lane & 15);
    const int nb   = kt*32 + ((lane & 16) >> 1);

    unsigned short vals[16];
    if (col < NH*DH + NH) {
        const int h = (col < NH*DH) ? (col >> 5) : (col - NH*DH);
        const int b = i*NH + h;
        const float cc = c1[b];
        #pragma unroll
        for (int k = 0; k < 16; ++k) {
            const int n = nb + (k & 7) + ((k & 8) << 1);
            const float s = __expf(a_src[(size_t)b*NN + n] - cc);
            const float v = (col < NH*DH)
                ? s * fp[((size_t)b*NN + n)*DH + (col & 31)] : s;
            vals[k] = f32_to_bf16(v);
        }
    } else {
        #pragma unroll
        for (int k = 0; k < 16; ++k) vals[k] = 0;
    }
    uint4* dst = (uint4*)(Bfrag + (size_t)g*16);
    dst[0] = *(uint4*)&vals[0];
    dst[1] = *(uint4*)&vals[8];
}

// ---------------------------------------------------------------------------
// Kernel 4: mask-GEMM  C[i](3072x272) = adj[i]^T @ B[i].
// WG = 8 waves, tile 128 M x 96 N (6 col-tiles), K-step 32, double-buffered
// LDS, one barrier/iter.
// A: 4x global_load_b128 (imm offsets); adj in {0,1} so bf16 pack of a row
//    pair is (a0 | a1<<16) * 0x3F80 -> ds_store_2addr_b32 (kt-invariant).
// B: identity copy via global_load_async_to_lds_b128 + s_wait_asynccnt.
// sched_barrier keeps all 14 frag ds_load_b128s ahead of the WMMA burst so
// the waitcnt pass can stagger s_wait_dscnt instead of 0x0 per WMMA.
// ---------------------------------------------------------------------------
__global__ __launch_bounds__(256) void gat_gemm(
    const int* __restrict__ adj0, const int* __restrict__ adj1,
    const unsigned short* __restrict__ Bfrag, float* __restrict__ Cmat)
{
    __shared__ __attribute__((aligned(16))) unsigned int Alds[2][2048]; // 2x8KB
    __shared__ __attribute__((aligned(16))) unsigned int Blds[2][1536]; // 2x6KB

    const int i = blockIdx.z;
    const int* __restrict__ adj = i ? adj1 : adj0;
    float* __restrict__ Cm = Cmat + (size_t)i*NN*BC;
    const int m0   = blockIdx.y * 128;
    const int cg   = blockIdx.x;            // col group: ctf0 = cg*CTPG
    const int tid  = threadIdx.x;
    const int lane = tid & 31;
    const int w    = tid >> 5;

    // --- kt-invariant addressing -------------------------------------------
    // A: thread owns row-pairs kk0 = 2*(tid>>5) + 16q (q=0,1), cols mm4 = lane*4
    const char* aptr = (const char*)(adj + (size_t)(2*(tid >> 5))*NN + m0 + (lane << 2));
    unsigned asto[2][4];
    #pragma unroll
    for (int q = 0; q < 2; ++q) {
        const int kk0 = 2*(tid >> 5) + 16*q;
        const int vv  = ((kk0 >> 4) << 2) + ((kk0 & 7) >> 1);
        #pragma unroll
        for (int x = 0; x < 4; ++x) {
            const int mm = (lane << 2) + x;
            const int s  = mm >> 4;
            const int ln = (mm & 15) + (((kk0 >> 3) & 1) << 4);
            asto[q][x] = (unsigned)((s*32 + ln)*8 + vv);
        }
    }
    // B: per-kt contiguous 6KB slab (6 col-tiles); 256x16B + 128x16B asyncs
    const unsigned short* bslab = Bfrag + ((size_t)(i*NKT)*NCT + cg*CTPG)*512;
    const unsigned voff = (unsigned)(tid * 16);
    unsigned bdst[2];
    bdst[0] = (unsigned)(size_t)&Blds[0][tid*4];
    bdst[1] = (unsigned)(size_t)&Blds[1][tid*4];

#define ISSUE_B(buf_, kt_) do {                                              \
        const unsigned short* sb_ = bslab + (size_t)(kt_)*(NCT*512);         \
        asm volatile("global_load_async_to_lds_b128 %0, %1, %2 offset:0"     \
                     :: "v"(bdst[buf_]), "v"(voff), "s"(sb_) : "memory");    \
        if (w < 4)                                                           \
            asm volatile("global_load_async_to_lds_b128 %0, %1, %2 offset:0" \
                         :: "v"(bdst[buf_] + 4096u), "v"(voff),              \
                            "s"(sb_ + 2048) : "memory");                     \
    } while (0)

    v8f zz = {0,0,0,0,0,0,0,0};
    v8f acc[CTPG] = {zz, zz, zz, zz, zz, zz};

    int4 ar[4];
#define LOAD_A(base) do {                              \
        ar[0] = *(const int4*)((base));                \
        ar[1] = *(const int4*)((base) + NN*4);         \
        ar[2] = *(const int4*)((base) + 16*NN*4);      \
        ar[3] = *(const int4*)((base) + 17*NN*4);      \
    } while (0)

    // --- prologue: tile 0 in flight ----------------------------------------
    LOAD_A(aptr);
    ISSUE_B(0, 0);

    for (int kt = 0; kt < NKT; ++kt) {
        const int buf = kt & 1;
        // pack + store A(kt): rows in {0,1} -> (a0 | a1<<16) * 0x3F80
        #pragma unroll
        for (int q = 0; q < 2; ++q) {
            const unsigned* a0 = (const unsigned*)&ar[2*q];
            const unsigned* a1 = (const unsigned*)&ar[2*q + 1];
            #pragma unroll
            for (int x = 0; x < 4; ++x)
                Alds[buf][asto[q][x]] = (a0[x] | (a1[x] << 16)) * 0x3F80u;
        }
        asm volatile("s_wait_asynccnt 0x0" ::: "memory");  // B(kt) landed
        __syncthreads();
        if (kt < NKT - 1) {   // overlap next tile with this tile's math
            LOAD_A(aptr + (size_t)(kt + 1)*32*NN*4);
            ISSUE_B(buf ^ 1, kt + 1);
        }
        // fragment preloads (2x ds_load_b128 each) ...
        FragBF fa, fb[CTPG];
        const uint4* A4 = (const uint4*)&Alds[buf][0];
        const int ab = ((w << 5) + lane) << 1;
        fa.q[0] = A4[ab];
        fa.q[1] = A4[ab + 1];
        const uint4* B4 = (const uint4*)&Blds[buf][0];
        #pragma unroll
        for (int ct = 0; ct < CTPG; ++ct) {
            const int bb = ((ct << 5) + lane) << 1;
            fb[ct].q[0] = B4[bb];
            fb[ct].q[1] = B4[bb + 1];
        }
        // ... keep every ds_load issued before the WMMA burst
        __builtin_amdgcn_sched_barrier(0);
        #pragma unroll
        for (int ct = 0; ct < CTPG; ++ct)
            acc[ct] = __builtin_amdgcn_wmma_f32_16x16x32_bf16(
                false, fa.v, false, fb[ct].v, (short)0, acc[ct], false, false);
    }
#undef LOAD_A
#undef ISSUE_B

    // --- store C; D layout: VGPR r -> M = r + 8*(lane>=16), N = lane&15 ----
    const int mrow = m0 + (w << 4) + ((lane >> 4) << 3);
    #pragma unroll
    for (int ct = 0; ct < CTPG; ++ct) {
        const int col = cg*(CTPG*16) + (ct << 4) + (lane & 15);
        if (col < BC) {
            #pragma unroll
            for (int r = 0; r < 8; ++r)
                Cm[(size_t)(mrow + r)*BC + col] = acc[ct][r];
        }
    }
}

// ---------------------------------------------------------------------------
// Kernel 5: out[s][n][h*32+e] = C[1-s][n][h*32+e] / (C[1-s][n][256+h] + 1e-9)
// ---------------------------------------------------------------------------
__global__ __launch_bounds__(256) void finalize_kernel(
    const float* __restrict__ Cmat, float* __restrict__ out)
{
    const int b = blockIdx.x;          // 0..2*NN-1
    const int s = b / NN;              // output node-set
    const int n = b - s*NN;
    const int i = 1 - s;               // edge type targeting set s
    const int j = threadIdx.x;         // 0..255
    const float* Cr = Cmat + ((size_t)i*NN + n)*BC;
    const float num = Cr[j];
    const float den = Cr[NH*DH + (j >> 5)];
    out[((size_t)s*NN + n)*(NH*DH) + j] = num / (den + 1e-9f);
}

// ---------------------------------------------------------------------------
extern "C" void kernel_launch(void* const* d_in, const int* in_sizes, int n_in,
                              void* d_out, int out_size, void* d_ws, size_t ws_size,
                              hipStream_t stream)
{
    (void)in_sizes; (void)n_in; (void)out_size; (void)ws_size;
    const float* emb0 = (const float*)d_in[0];
    const float* emb1 = (const float*)d_in[1];
    const int*   adj0 = (const int*)d_in[2];
    const int*   adj1 = (const int*)d_in[3];
    const float* W    = (const float*)d_in[4];
    const float* attn = (const float*)d_in[5];
    float* out = (float*)d_out;

    char* ws = (char*)d_ws;
    size_t off = 0;
    float* fp    = (float*)(ws + off);  off += (size_t)2*NH*NN*DH*4;            // 6.29 MB
    float* a_src = (float*)(ws + off);  off += (size_t)2*NH*NN*4;               // 197 KB
    float* c1    = (float*)(ws + off);  off += 1024;
    unsigned short* Bfrag = (unsigned short*)(ws + off);
    off += (size_t)2*NKT*NCT*32*16*2;                                           // 3.54 MB
    float* Cmat  = (float*)(ws + off);                                          // 6.68 MB

    proj_kernel<<<dim3(2, NH, NN/16), 256, 0, stream>>>(emb0, emb1, W, attn, fp, a_src);
    max_kernel<<<2*NH, 256, 0, stream>>>(a_src, c1);
    bmat_kernel<<<(2*NKT*NCT*32)/256, 256, 0, stream>>>(fp, a_src, c1, Bfrag);
    gat_gemm<<<dim3(3, NN/128, 2), 256, 0, stream>>>(adj0, adj1, Bfrag, Cmat);
    finalize_kernel<<<2*NN, 256, 0, stream>>>(Cmat, out);
}
